// TransformerBlock_15290083573798
// MI455X (gfx1250) — matrix operand order repaired and checked by reading, not compiled, not run on hardware
//
#include <hip/hip_runtime.h>

// ---------------------------------------------------------------------------
// MI455X (gfx1250) transformer block, bf16 WMMA everywhere.
//  - fp32 -> bf16 conversion once per tensor (v_cvt_pk_bf16_f32)
//  - GEMM + attention-V tiles staged with global_load_async_to_lds_b128
//    (ASYNCcnt), double-buffered in the GEMM so DMA overlaps v_wmma
//  - flash attention: QK^T and P*V on v_wmma_f32_16x16x32_bf16; softmax
//    row reductions via DPP16 lane permutes (no LDS latency)
// ---------------------------------------------------------------------------

typedef __bf16 bf16x16 __attribute__((ext_vector_type(16)));
typedef __bf16 bf16x2  __attribute__((ext_vector_type(2)));
typedef float  f32x8   __attribute__((ext_vector_type(8)));
typedef float  f32x2   __attribute__((ext_vector_type(2)));

union FragU {
    bf16x16        v;
    uint4          u4[2];
    unsigned short us[16];
};

__device__ __forceinline__ unsigned cvt2(float a, float b) {
    f32x2 f; f[0] = a; f[1] = b;
    bf16x2 h = __builtin_convertvector(f, bf16x2);
    return __builtin_bit_cast(unsigned, h);
}
__device__ __forceinline__ unsigned short f2bf_hw(float x) {
    __bf16 h = (__bf16)x;
    return __builtin_bit_cast(unsigned short, h);
}

#define WMMA_BF16(A, B, C) \
    __builtin_amdgcn_wmma_f32_16x16x32_bf16(false, (A), false, (B), (short)0, (C), false, false)

__device__ __forceinline__ unsigned lds_addr32(const void* p) {
    return (unsigned)(uintptr_t)p;   // low 32 bits of shared-aperture flat addr
}

// ---- DPP16 helpers: reduce over the 16-lane rows of a wave32 --------------
template <int CTRL>
__device__ __forceinline__ float dppf(float x) {
    int i = __builtin_bit_cast(int, x);
    i = __builtin_amdgcn_update_dpp(0, i, CTRL, 0xF, 0xF, true);
    return __builtin_bit_cast(float, i);
}
#define DPP_XOR1   0xB1   // quad_perm [1,0,3,2]
#define DPP_XOR2   0x4E   // quad_perm [2,3,0,1]
#define DPP_HMIRR  0x141  // row_half_mirror
#define DPP_MIRR   0x140  // row_mirror

__device__ __forceinline__ float row_max16(float x) {
    x = fmaxf(x, dppf<DPP_XOR1>(x));
    x = fmaxf(x, dppf<DPP_XOR2>(x));
    x = fmaxf(x, dppf<DPP_HMIRR>(x));
    x = fmaxf(x, dppf<DPP_MIRR>(x));
    return x;
}
__device__ __forceinline__ float row_sum16(float x) {
    x += dppf<DPP_XOR1>(x);
    x += dppf<DPP_XOR2>(x);
    x += dppf<DPP_HMIRR>(x);   // other quad's (uniform) partial sum
    x += dppf<DPP_MIRR>(x);    // other half-row's (uniform) partial sum
    return x;
}

// ---------------------------------------------------------------------------
// fp32 -> bf16 bulk convert, 8 elements / thread
// ---------------------------------------------------------------------------
__global__ void f32_to_bf16_kernel(const float* __restrict__ in,
                                   unsigned short* __restrict__ out, int n8) {
    int i = blockIdx.x * blockDim.x + threadIdx.x;
    if (i >= n8) return;
    const float4* p = (const float4*)in + (size_t)i * 2;
    float4 a = p[0], b = p[1];
    uint4 r;
    r.x = cvt2(a.x, a.y); r.y = cvt2(a.z, a.w);
    r.z = cvt2(b.x, b.y); r.w = cvt2(b.z, b.w);
    ((uint4*)out)[i] = r;
}

// ---------------------------------------------------------------------------
// GEMM: C[M,N] = A[M,K] @ W[N,K]^T, bf16 in, fp32 accum.
// 8 waves, block tile 128x128, wave tile 32x64, K step 32.
// Async global->LDS b128 staging, double buffered.
// ---------------------------------------------------------------------------
template <bool OUT_BF16>
__global__ __launch_bounds__(256)
void gemm_bf16(const unsigned short* __restrict__ A, const unsigned short* __restrict__ W,
               void* __restrict__ Cv, int M, int N, int K) {
    __shared__ alignas(16) unsigned short As[2][128 * 40];
    __shared__ alignas(16) unsigned short Bs[2][128 * 40];

    const int tid  = threadIdx.x;
    const int lane = tid & 31;
    const int wave = tid >> 5;
    const int wm   = wave >> 1;
    const int wn   = wave & 1;
    const int bm   = blockIdx.y * 128;
    const int bn   = blockIdx.x * 128;

    const unsigned long long abase = (unsigned long long)(uintptr_t)A;
    const unsigned long long wbase = (unsigned long long)(uintptr_t)W;
    const unsigned asb[2] = {lds_addr32(&As[0][0]), lds_addr32(&As[1][0])};
    const unsigned bsb[2] = {lds_addr32(&Bs[0][0]), lds_addr32(&Bs[1][0])};

    f32x8 acc[2][4];
#pragma unroll
    for (int i = 0; i < 2; ++i)
#pragma unroll
        for (int j = 0; j < 4; ++j) acc[i][j] = (f32x8)0.0f;

    const int khalf = (lane >= 16) ? 8 : 0;
    const int kb    = (lane >= 16) ? 16 : 0;
    const int ntiles = K >> 5;

    auto issue = [&](int kt, int buf) {
#pragma unroll
        for (int it = 0; it < 2; ++it) {
            int idx = it * 256 + tid;             // 0..511
            int row = idx >> 2;                   // 0..127
            int c   = idx & 3;                    // 16B chunk
            unsigned lo  = (unsigned)(row * 80 + c * 16);
            unsigned goa = (unsigned)((((size_t)(bm + row)) * K + kt * 32) * 2 + c * 16);
            unsigned gob = (unsigned)((((size_t)(bn + row)) * K + kt * 32) * 2 + c * 16);
            asm volatile("global_load_async_to_lds_b128 %0, %1, %2"
                         :: "v"(asb[buf] + lo), "v"(goa), "s"(abase) : "memory");
            asm volatile("global_load_async_to_lds_b128 %0, %1, %2"
                         :: "v"(bsb[buf] + lo), "v"(gob), "s"(wbase) : "memory");
        }
    };

    issue(0, 0);

    for (int kt = 0; kt < ntiles; ++kt) {
        const int cur = kt & 1;
        asm volatile("s_wait_asynccnt 0x0" ::: "memory");
        __syncthreads();
        if (kt + 1 < ntiles) issue(kt + 1, cur ^ 1);

        const unsigned short* Al = As[cur];
        const unsigned short* Bl = Bs[cur];

        bf16x16 af[2], bfr[4];
#pragma unroll
        for (int i = 0; i < 2; ++i) {
            int m = wm * 32 + i * 16 + (lane & 15);
            FragU fu;
            fu.u4[0] = *(const uint4*)(&Al[m * 40 + khalf]);
            fu.u4[1] = *(const uint4*)(&Al[m * 40 + 16 + khalf]);
            af[i] = fu.v;
        }
#pragma unroll
        for (int j = 0; j < 4; ++j) {
            int n = wn * 64 + j * 16 + (lane & 15);
            FragU fu;
            fu.u4[0] = *(const uint4*)(&Bl[n * 40 + khalf]);
            fu.u4[1] = *(const uint4*)(&Bl[n * 40 + 16 + khalf]);
            bfr[j] = fu.v;
        }
#pragma unroll
        for (int i = 0; i < 2; ++i)
#pragma unroll
            for (int j = 0; j < 4; ++j)
                acc[i][j] = WMMA_BF16(af[i], bfr[j], acc[i][j]);
    }

#pragma unroll
    for (int i = 0; i < 2; ++i)
#pragma unroll
        for (int j = 0; j < 4; ++j) {
            int n  = bn + wn * 64 + j * 16 + (lane & 15);
            int m0 = bm + wm * 32 + i * 16 + ((lane >= 16) ? 8 : 0);
            if constexpr (OUT_BF16) {
                unsigned short* Cb = (unsigned short*)Cv;
#pragma unroll
                for (int r = 0; r < 8; ++r)
                    Cb[(size_t)(m0 + r) * N + n] = f2bf_hw(acc[i][j][r]);
            } else {
                float* Cf = (float*)Cv;
#pragma unroll
                for (int r = 0; r < 8; ++r)
                    Cf[(size_t)(m0 + r) * N + n] = acc[i][j][r];
            }
        }
}

// ---------------------------------------------------------------------------
// RoPE: fp32 q,k -> rotated bf16. 1 thread = 1 even/odd pair.
// ---------------------------------------------------------------------------
__global__ void rope_kernel(const float* __restrict__ q, const float* __restrict__ k,
                            unsigned short* __restrict__ qb, unsigned short* __restrict__ kb) {
    int idx = blockIdx.x * blockDim.x + threadIdx.x;
    if (idx >= 4096 * 16 * 32) return;
    int pair = idx & 31;
    int h    = (idx >> 5) & 15;
    int row  = idx >> 9;
    int s    = row & 2047;
    float inv = __powf(10000.0f, -(float)pair * (1.0f / 32.0f));
    float sn, cs;
    __sincosf((float)s * inv, &sn, &cs);
    size_t off = (size_t)row * 1024 + h * 64 + pair * 2;
    float qe = q[off], qo = q[off + 1];
    float ke = k[off], ko = k[off + 1];
    *(unsigned*)(qb + off) = cvt2(qe * cs - qo * sn, qo * cs + qe * sn);
    *(unsigned*)(kb + off) = cvt2(ke * cs - ko * sn, ko * cs + ke * sn);
}

// ---------------------------------------------------------------------------
// Flash attention, bf16 in/out. 4 waves/block, wave = 16 query rows.
// Unmasked hot loop + one masked diagonal block. V staged via async DMA.
// ---------------------------------------------------------------------------
__global__ __launch_bounds__(128)
void attn_kernel(const unsigned short* __restrict__ Qb, const unsigned short* __restrict__ Kb,
                 const unsigned short* __restrict__ Vb, unsigned short* __restrict__ Ob) {
    __shared__ alignas(16) unsigned short p_lds[4][16 * 32];
    __shared__ alignas(16) unsigned short v_lds[4][32 * 80];   // 32 keys x 64 d, 160B rows

    const int lane = threadIdx.x & 31;
    const int wave = threadIdx.x >> 5;
    const int bh   = blockIdx.y;
    const int b    = bh >> 4;
    const int h    = bh & 15;
    const int qg0  = (blockIdx.x * 4 + wave) * 16;

    const size_t headoff = (size_t)b * 2048 * 1024 + (size_t)h * 64;
    const int khalf = (lane >= 16) ? 8 : 0;
    const int kbse  = (lane >= 16) ? 16 : 0;
    const int nl    = lane & 15;
    const int rbase = (lane >= 16) ? 8 : 0;

    const unsigned long long vbase = (unsigned long long)(uintptr_t)Vb;
    unsigned short* pw = p_lds[wave];
    unsigned short* vw = v_lds[wave];
    const unsigned vwb = lds_addr32(vw);

    bf16x16 qf[2];
#pragma unroll
    for (int kc = 0; kc < 2; ++kc) {
        const unsigned short* qrow = Qb + headoff + (size_t)(qg0 + nl) * 1024 + kc * 32;
        FragU fu;
        fu.u4[0] = *(const uint4*)(qrow + khalf);
        fu.u4[1] = *(const uint4*)(qrow + 16 + khalf);
        qf[kc] = fu.v;
    }

    f32x8 oacc[4];
#pragma unroll
    for (int j = 0; j < 4; ++j) oacc[j] = (f32x8)0.0f;
    float m8[8], l8[8];
#pragma unroll
    for (int r = 0; r < 8; ++r) { m8[r] = -1e30f; l8[r] = 0.0f; }

    auto kblock = [&](int kg0, bool domask) {
        // ---- async DMA: V 32x64 -> per-wave LDS (overlaps QK wmma) ---------
#pragma unroll
        for (int i = 0; i < 8; ++i) {
            int ci  = i * 32 + lane;      // 0..255 chunks of 16B
            int row = ci >> 3;            // key 0..31
            int c   = ci & 7;             // 8-elem chunk in d
            unsigned lo = (unsigned)(row * 160 + c * 16);
            unsigned go = (unsigned)((headoff + (size_t)(kg0 + row) * 1024 + c * 8) * 2);
            asm volatile("global_load_async_to_lds_b128 %0, %1, %2"
                         :: "v"(vwb + lo), "v"(go), "s"(vbase) : "memory");
        }

        // ---- scores: two 16x16 tiles ---------------------------------------
        f32x8 s[2];
#pragma unroll
        for (int t = 0; t < 2; ++t) {
            s[t] = (f32x8)0.0f;
            int key = kg0 + t * 16 + nl;
#pragma unroll
            for (int kc = 0; kc < 2; ++kc) {
                const unsigned short* krow = Kb + headoff + (size_t)key * 1024 + kc * 32 + khalf;
                FragU fu;
                fu.u4[0] = *(const uint4*)(krow);
                fu.u4[1] = *(const uint4*)(krow + 16);
                s[t] = WMMA_BF16(qf[kc], fu.v, s[t]);
            }
            if (domask) {
#pragma unroll
                for (int r = 0; r < 8; ++r) {
                    int qg = qg0 + r + rbase;
                    int kg = kg0 + t * 16 + nl;
                    float sv = s[t][r] * 0.125f;
                    s[t][r] = (kg <= qg) ? sv : -1e30f;
                }
            } else {
#pragma unroll
                for (int r = 0; r < 8; ++r) s[t][r] *= 0.125f;
            }
        }

        // ---- online softmax: DPP16 row reductions --------------------------
        float alpha[8];
#pragma unroll
        for (int r = 0; r < 8; ++r) {
            float mx = row_max16(fmaxf(s[0][r], s[1][r]));
            float mn = fmaxf(m8[r], mx);
            alpha[r] = __expf(m8[r] - mn);
            m8[r] = mn;
        }
#pragma unroll
        for (int t = 0; t < 2; ++t)
#pragma unroll
            for (int r = 0; r < 8; ++r)
                s[t][r] = __expf(s[t][r] - m8[r]);

#pragma unroll
        for (int r = 0; r < 8; ++r) {
            float rs = row_sum16(s[0][r] + s[1][r]);
            l8[r] = l8[r] * alpha[r] + rs;
        }

#pragma unroll
        for (int j = 0; j < 4; ++j)
#pragma unroll
            for (int r = 0; r < 8; ++r) oacc[j][r] *= alpha[r];

        // ---- P: C layout -> LDS -> A layout --------------------------------
#pragma unroll
        for (int t = 0; t < 2; ++t)
#pragma unroll
            for (int r = 0; r < 8; ++r)
                pw[(r + rbase) * 32 + t * 16 + nl] = f2bf_hw(s[t][r]);

        asm volatile("s_wait_dscnt 0x0" ::: "memory");    // P tile visible
        asm volatile("s_wait_asynccnt 0x0" ::: "memory"); // V tile resident

        FragU pf;
        pf.u4[0] = *(const uint4*)(&pw[nl * 32 + kbse]);
        pf.u4[1] = *(const uint4*)(&pw[nl * 32 + kbse + 8]);

        // ---- O += P @ V ----------------------------------------------------
#pragma unroll
        for (int j = 0; j < 4; ++j) {
            int d = j * 16 + nl;
            FragU vf;
#pragma unroll
            for (int c = 0; c < 16; ++c)
                vf.us[c] = vw[(kbse + c) * 80 + d];
            oacc[j] = WMMA_BF16(pf.v, vf.v, oacc[j]);
        }
    };

    const int nfull = qg0 >> 5;                 // fully-unmasked key blocks
    for (int kb2 = 0; kb2 < nfull; ++kb2) kblock(kb2 * 32, false);
    kblock(nfull * 32, true);                   // single diagonal block

#pragma unroll
    for (int j = 0; j < 4; ++j)
#pragma unroll
        for (int r = 0; r < 8; ++r)
            Ob[headoff + (size_t)(qg0 + r + rbase) * 1024 + j * 16 + nl] =
                f2bf_hw(oacc[j][r] / l8[r]);
}

// ---------------------------------------------------------------------------
// hb = bf16( silu(g) * u )
// ---------------------------------------------------------------------------
__global__ void silu_mul_kernel(const float* __restrict__ g, const float* __restrict__ u,
                                unsigned short* __restrict__ hb, int n2) {
    int i = blockIdx.x * blockDim.x + threadIdx.x;
    if (i >= n2) return;
    float g0 = g[2 * (size_t)i], g1 = g[2 * (size_t)i + 1];
    float u0 = u[2 * (size_t)i], u1 = u[2 * (size_t)i + 1];
    float h0 = g0 * (1.0f / (1.0f + __expf(-g0))) * u0;
    float h1 = g1 * (1.0f / (1.0f + __expf(-g1))) * u1;
    *(unsigned*)(hb + 2 * (size_t)i) = cvt2(h0, h1);
}

// ---------------------------------------------------------------------------
// out = x + rmsnorm(y)*gain ; optional bf16 mirror.
// ---------------------------------------------------------------------------
__global__ __launch_bounds__(256)
void resid_rmsnorm_kernel(const float* __restrict__ x, const float* __restrict__ y,
                          const float* __restrict__ gain, float* __restrict__ out,
                          unsigned short* __restrict__ outb) {
    __shared__ float sred[8];
    const int row = blockIdx.x;
    const float* yr = y + (size_t)row * 1024;
    const float* xr = x + (size_t)row * 1024;

    float ss = 0.0f;
    for (int c = threadIdx.x; c < 1024; c += 256) { float v = yr[c]; ss += v * v; }
#pragma unroll
    for (int off = 1; off < 32; off <<= 1) ss += __shfl_xor(ss, off, 32);
    if ((threadIdx.x & 31) == 0) sred[threadIdx.x >> 5] = ss;
    __syncthreads();
    float total = 0.0f;
#pragma unroll
    for (int i = 0; i < 8; ++i) total += sred[i];
    float scale = rsqrtf(total * (1.0f / 1024.0f) + 1e-5f);
    for (int c = threadIdx.x; c < 1024; c += 256) {
        float o = xr[c] + yr[c] * scale * gain[c];
        out[(size_t)row * 1024 + c] = o;
        if (outb) outb[(size_t)row * 1024 + c] = f2bf_hw(o);
    }
}

// ---------------------------------------------------------------------------
// Orchestration
// ---------------------------------------------------------------------------
extern "C" void kernel_launch(void* const* d_in, const int* in_sizes, int n_in,
                              void* d_out, int out_size, void* d_ws, size_t ws_size,
                              hipStream_t stream) {
    const float* x         = (const float*)d_in[0];
    const float* w_q       = (const float*)d_in[1];
    const float* w_k       = (const float*)d_in[2];
    const float* w_v       = (const float*)d_in[3];
    const float* w_o       = (const float*)d_in[4];
    const float* attn_gain = (const float*)d_in[5];
    const float* ffn_gain  = (const float*)d_in[6];
    const float* w1        = (const float*)d_in[7];
    const float* w2        = (const float*)d_in[8];
    const float* w3        = (const float*)d_in[9];
    float* out = (float*)d_out;

    typedef unsigned short ush;
    const size_t E_XD = (size_t)4096 * 1024;
    const size_t E_W  = (size_t)1024 * 1024;
    const size_t E_WF = (size_t)4096 * 1024;
    const size_t E_FF = (size_t)4096 * 4096;

    char* ws = (char*)d_ws;
    size_t o = 0;
    auto carve = [&](size_t bytes) { void* p = ws + o; o += (bytes + 255) & ~(size_t)255; return p; };

    ush*   xb    = (ush*)carve(E_XD * 2);
    ush*   wqb   = (ush*)carve(E_W * 2);
    ush*   wkb   = (ush*)carve(E_W * 2);
    ush*   wvb   = (ush*)carve(E_W * 2);
    ush*   wob   = (ush*)carve(E_W * 2);
    ush*   w1b   = (ush*)carve(E_WF * 2);
    ush*   w3b   = (ush*)carve(E_WF * 2);
    ush*   w2b   = (ush*)carve(E_WF * 2);
    float* q     = (float*)carve(E_XD * 4);
    float* k     = (float*)carve(E_XD * 4);
    ush*   qb    = (ush*)carve(E_XD * 2);
    ush*   kb    = (ush*)carve(E_XD * 2);
    ush*   vb    = (ush*)carve(E_XD * 2);
    ush*   attnb = (ush*)carve(E_XD * 2);
    float* x1    = (float*)carve(E_XD * 4);
    ush*   x1b   = (ush*)carve(E_XD * 2);
    float* g     = (float*)carve(E_FF * 4);
    float* u     = (float*)carve(E_FF * 4);
    ush*   hb    = (ush*)carve(E_FF * 2);
    float* proj  = q;                              // q fp32 dead after rope
    float* ffn   = k;                              // k fp32 dead after rope

    const dim3 blk256(256), blk128(128);
    const dim3 grid_1k(1024 / 128, 4096 / 128);
    const dim3 grid_4k(4096 / 128, 4096 / 128);
    auto cvtN = [&](const float* src, ush* dst, size_t n) {
        int n8 = (int)(n / 8);
        f32_to_bf16_kernel<<<(n8 + 255) / 256, blk256, 0, stream>>>(src, dst, n8);
    };

    cvtN(x, xb, E_XD);
    cvtN(w_q, wqb, E_W);  cvtN(w_k, wkb, E_W);
    cvtN(w_v, wvb, E_W);  cvtN(w_o, wob, E_W);
    cvtN(w1, w1b, E_WF);  cvtN(w3, w3b, E_WF);  cvtN(w2, w2b, E_WF);

    gemm_bf16<false><<<grid_1k, blk256, 0, stream>>>(xb, wqb, q,  4096, 1024, 1024);
    gemm_bf16<false><<<grid_1k, blk256, 0, stream>>>(xb, wkb, k,  4096, 1024, 1024);
    gemm_bf16<true ><<<grid_1k, blk256, 0, stream>>>(xb, wvb, vb, 4096, 1024, 1024);

    rope_kernel<<<(4096 * 16 * 32) / 256, blk256, 0, stream>>>(q, k, qb, kb);

    attn_kernel<<<dim3(32, 32), blk128, 0, stream>>>(qb, kb, vb, attnb);

    gemm_bf16<false><<<grid_1k, blk256, 0, stream>>>(attnb, wob, proj, 4096, 1024, 1024);
    resid_rmsnorm_kernel<<<4096, blk256, 0, stream>>>(x, proj, attn_gain, x1, x1b);

    gemm_bf16<false><<<grid_4k, blk256, 0, stream>>>(x1b, w1b, g, 4096, 4096, 1024);
    gemm_bf16<false><<<grid_4k, blk256, 0, stream>>>(x1b, w3b, u, 4096, 4096, 1024);
    silu_mul_kernel<<<(int)(E_FF / 2 / 256), blk256, 0, stream>>>(g, u, hb, (int)(E_FF / 2));
    gemm_bf16<false><<<grid_1k, blk256, 0, stream>>>(hb, w2b, ffn, 4096, 1024, 4096);

    resid_rmsnorm_kernel<<<4096, blk256, 0, stream>>>(x1, ffn, ffn_gain, out, (ush*)nullptr);
}
